// SSMBlock_16320875724888
// MI455X (gfx1250) — matrix operand order repaired and checked by reading, not compiled, not run on hardware
//
#include <hip/hip_runtime.h>
#include <hip/hip_bf16.h>
#include <stdint.h>

// ---------------------------------------------------------------------------
// Model dims (compile-time constants from the reference)
// ---------------------------------------------------------------------------
#define HID     2560
#define D_INNER 8192
#define NG      32
#define D_SSM   4096
#define CONV_K  4
#define HD      128
#define DS      64
#define SEQ_L   1024
#define EPS_RMS 1e-6f

typedef __attribute__((ext_vector_type(16))) __bf16 v16bf;
typedef __attribute__((ext_vector_type(8)))  float  v8f;
typedef __attribute__((ext_vector_type(4)))  int    v4i_t;

union FragBF {
    uint32_t u[8];
    v16bf    v;
};

#if defined(__has_builtin)
#if __has_builtin(__builtin_amdgcn_global_load_async_to_lds_b32) && \
    __has_builtin(__builtin_amdgcn_s_wait_asynccnt)
#define HAVE_ASYNC_LDS32 1
#endif
#if __has_builtin(__builtin_amdgcn_global_load_async_to_lds_b128) && \
    __has_builtin(__builtin_amdgcn_s_wait_asynccnt)
#define HAVE_ASYNC_LDS128 1
#endif
#endif

// ---- f32 -> bf16 helpers (manual RNE; HW cvt builtin absent on toolchain) --
__device__ __forceinline__ uint32_t pack_bf16x2(float a, float b) {
    uint32_t ua = __builtin_bit_cast(uint32_t, a);
    uint32_t ub = __builtin_bit_cast(uint32_t, b);
    ua = (ua + 0x7FFFu + ((ua >> 16) & 1u)) >> 16;
    ub = (ub + 0x7FFFu + ((ub >> 16) & 1u)) >> 16;
    return ua | (ub << 16);
}
__device__ __forceinline__ uint16_t bf16_of(float a) {
    uint32_t ua = __builtin_bit_cast(uint32_t, a);
    return (uint16_t)((ua + 0x7FFFu + ((ua >> 16) & 1u)) >> 16);
}

#ifdef HAVE_ASYNC_LDS32
__device__ __forceinline__ void async_ld_b32(const void* g, void* l) {
    __builtin_amdgcn_global_load_async_to_lds_b32(
        (__attribute__((address_space(1))) int*)(uintptr_t)g,
        (__attribute__((address_space(3))) int*)(uint32_t)(uintptr_t)l, 0, 0);
}
#endif
#ifdef HAVE_ASYNC_LDS128
__device__ __forceinline__ void async_ld_b128(const void* g, void* l) {
    __builtin_amdgcn_global_load_async_to_lds_b128(
        (__attribute__((address_space(1))) v4i_t*)(uintptr_t)g,
        (__attribute__((address_space(3))) v4i_t*)(uint32_t)(uintptr_t)l, 0, 0);
}
#endif

__device__ __forceinline__ float sigmoidf_(float x) { return 1.f / (1.f + __expf(-x)); }

// ---------------------------------------------------------------------------
// f32 -> bf16 bulk conversion (bandwidth-bound pre-pass); n % 4 == 0
// ---------------------------------------------------------------------------
__global__ __launch_bounds__(256)
void cvt_f32_bf16(const float* __restrict__ src, uint32_t* __restrict__ dst, int n4) {
    int i = blockIdx.x * 256 + threadIdx.x;
    if (i < n4) {
        float4 v = ((const float4*)src)[i];
        dst[i * 2]     = pack_bf16x2(v.x, v.y);
        dst[i * 2 + 1] = pack_bf16x2(v.z, v.w);
    }
}

// ---------------------------------------------------------------------------
// WMMA GEMM (pure bf16 inputs):  C[M,N] = A[M,K] * W[N,K]^T  (f32 accum/out)
// Block tile 256x128, 256 threads (8 wave32 waves), each wave -> 64x64 (4x4).
// Double-buffered LDS, tiles staged via GLOBAL_LOAD_ASYNC_TO_LDS_B128,
// 16x v_wmma_f32_16x16x32_bf16 per wave per K-step (1.0 ds_b128 per WMMA).
// Requires: M%256==0, N%128==0, K%32==0, K/32 >= 2.
// ---------------------------------------------------------------------------
__global__ __launch_bounds__(256)
void gemm_bf16_wmma(const uint16_t* __restrict__ A, const uint16_t* __restrict__ W,
                    float* __restrict__ C, int M, int N, int K) {
    __shared__ uint32_t As[2][256][20];   // [buf][row][16 dwords bf16 data + pad]
    __shared__ uint32_t Bs[2][128][20];

    const int tid  = threadIdx.x;
    const int lane = tid & 31;
    const int wid  = tid >> 5;
    const int wm   = wid >> 1;          // 0..3 -> M offset wm*64
    const int wn   = wid & 1;           // 0..1 -> N offset wn*64
    const int lrow = lane & 15;
    const int hi   = lane >> 4;         // half-wave select
    const int kbA  = hi * 4;            // A dword offset within 8-dword half
    const int pbB  = hi * 4;            // B dword offset

    const uint16_t* Ablk = A + (size_t)blockIdx.y * 256 * K;
    const uint16_t* Wblk = W + (size_t)blockIdx.x * 128 * K;

    // stage geometry: rows x 64B; 16B chunks; A: 1024 chunks, B: 512 chunks
    const int srow0 = tid >> 2;          // base row (0..63), +64 per iteration
    const int sq    = tid & 3;           // 16B chunk within row (4 chunks = 64B)

    const v8f vzero = {0.f, 0.f, 0.f, 0.f, 0.f, 0.f, 0.f, 0.f};
    v8f acc[4][4];
#pragma unroll
    for (int i = 0; i < 4; ++i)
#pragma unroll
        for (int j = 0; j < 4; ++j) acc[i][j] = vzero;

#ifdef HAVE_ASYNC_LDS128
    // ---- async global->LDS staging: A 256x32, B 128x32 (bf16) ----
    auto stage = [&](int k0, int buf) {
#pragma unroll
        for (int i = 0; i < 4; ++i) {
            int row = srow0 + i * 64;
            async_ld_b128(Ablk + (size_t)row * K + k0 + sq * 8, &As[buf][row][sq * 4]);
        }
#pragma unroll
        for (int i = 0; i < 2; ++i) {
            int row = srow0 + i * 64;
            async_ld_b128(Wblk + (size_t)row * K + k0 + sq * 8, &Bs[buf][row][sq * 4]);
        }
    };
#else
    int4 sa[4], sb[2];
    auto g2r = [&](int k0) {
#pragma unroll
        for (int i = 0; i < 4; ++i) {
            int row = srow0 + i * 64;
            sa[i] = *(const int4*)(Ablk + (size_t)row * K + k0 + sq * 8);
        }
#pragma unroll
        for (int i = 0; i < 2; ++i) {
            int row = srow0 + i * 64;
            sb[i] = *(const int4*)(Wblk + (size_t)row * K + k0 + sq * 8);
        }
    };
    auto r2l = [&](int buf) {
#pragma unroll
        for (int i = 0; i < 4; ++i) {
            int row = srow0 + i * 64;
            *(int4*)&As[buf][row][sq * 4] = sa[i];
        }
#pragma unroll
        for (int i = 0; i < 2; ++i) {
            int row = srow0 + i * 64;
            *(int4*)&Bs[buf][row][sq * 4] = sb[i];
        }
    };
#endif

    const int nk = K >> 5;
#ifdef HAVE_ASYNC_LDS128
    stage(0, 0);
    __builtin_amdgcn_s_wait_asynccnt(0);
    __syncthreads();
#else
    g2r(0);
    r2l(0);
    __syncthreads();
#endif

    for (int kt = 0; kt < nk; ++kt) {
        const int cur = kt & 1;
#ifdef HAVE_ASYNC_LDS128
        if (kt + 1 < nk) stage((kt + 1) << 5, cur ^ 1);  // lands during compute
#else
        if (kt + 1 < nk) g2r((kt + 1) << 5);
#endif

        // ---- A fragments (16x32 bf16, documented lane/VGPR layout) ----
        v16bf afr[4];
#pragma unroll
        for (int mt = 0; mt < 4; ++mt) {
            const uint32_t* Ar = &As[cur][wm * 64 + mt * 16 + lrow][0];
            FragBF fa;
#pragma unroll
            for (int j = 0; j < 4; ++j) fa.u[j]     = Ar[kbA + j];
#pragma unroll
            for (int j = 0; j < 4; ++j) fa.u[4 + j] = Ar[8 + kbA + j];
            afr[mt] = fa.v;
        }
        // ---- B fragments (32x16 bf16: lanes 0-15 K=0..7,16..23, 16-31 K=8..15,24..31) ----
        v16bf bfr[4];
#pragma unroll
        for (int nt = 0; nt < 4; ++nt) {
            const uint32_t* Br = &Bs[cur][wn * 64 + nt * 16 + lrow][0];
            FragBF fb;
#pragma unroll
            for (int j = 0; j < 8; ++j) fb.u[j] = Br[pbB + j + (j & 4)];
            bfr[nt] = fb.v;
        }

        // ---- 16 WMMAs per wave per K-step ----
#pragma unroll
        for (int mt = 0; mt < 4; ++mt)
#pragma unroll
            for (int nt = 0; nt < 4; ++nt)
                acc[mt][nt] = __builtin_amdgcn_wmma_f32_16x16x32_bf16(
                    false, afr[mt], false, bfr[nt], (short)0, acc[mt][nt],
                    false, false);

        if (kt + 1 < nk) {
#ifdef HAVE_ASYNC_LDS128
            __builtin_amdgcn_s_wait_asynccnt(0);
#else
            r2l(cur ^ 1);
#endif
            __syncthreads();
        }
    }

    // ---- epilogue: C/D layout -> lane lrow = N col, VGPR v = M row (+8 hi) ----
    // 32-bit element offsets (all outputs < 2^31 elements)
    float* Cblk = C + (size_t)blockIdx.y * 256 * N + (size_t)blockIdx.x * 128;
#pragma unroll
    for (int mt = 0; mt < 4; ++mt) {
#pragma unroll
        for (int nt = 0; nt < 4; ++nt) {
            int off = (wm * 64 + mt * 16 + hi * 8) * N + wn * 64 + nt * 16 + lrow;
#pragma unroll
            for (int v = 0; v < 8; ++v)
                Cblk[off + v * N] = acc[mt][nt][v];
        }
    }
}

// ---------------------------------------------------------------------------
// dt = h @ W_b.T, D = h @ W_a.T  (N=32 each -> one wave per output, shfl reduce)
// ---------------------------------------------------------------------------
__global__ __launch_bounds__(256)
void proj_small(const float* __restrict__ h, const float* __restrict__ Wa,
                const float* __restrict__ Wb, float* __restrict__ dt,
                float* __restrict__ Dv) {
    const int l    = blockIdx.x;
    const int lane = threadIdx.x & 31;
    const int wid  = threadIdx.x >> 5;
    const float* hr = h + (size_t)l * HID;
    for (int it = 0; it < 8; ++it) {
        int o = it * 8 + wid;                 // 0..63
        const float* wr = (o < NG) ? (Wb + (size_t)o * HID)
                                   : (Wa + (size_t)(o - NG) * HID);
        float s = 0.f;
        for (int k = lane; k < HID; k += 32) s += hr[k] * wr[k];
#pragma unroll
        for (int off = 16; off; off >>= 1) s += __shfl_xor(s, off, 32);
        if (lane == 0) {
            if (o < NG) dt[(size_t)l * NG + o] = s;
            else        Dv[(size_t)l * NG + (o - NG)] = s;
        }
    }
}

// ---------------------------------------------------------------------------
// causal depthwise conv1d (K=4) + SiLU
// ---------------------------------------------------------------------------
__global__ __launch_bounds__(256)
void conv_silu(const float* __restrict__ xBC, const float* __restrict__ w,
               float* __restrict__ out) {
    const int c = blockIdx.x * 256 + threadIdx.x;
    const int l = blockIdx.y;
    float acc = 0.f;
#pragma unroll
    for (int j = 0; j < CONV_K; ++j) {
        int ls = l + j - (CONV_K - 1);
        if (ls >= 0) acc += w[c * CONV_K + j] * xBC[(size_t)ls * D_INNER + c];
    }
    out[(size_t)l * D_INNER + c] = acc * sigmoidf_(acc);
}

// ---------------------------------------------------------------------------
// Sequential SSM scan: one workgroup per group g; thread owns (hd, 32 states).
// B/C rows staged to LDS each step (async global->LDS when available).
// ---------------------------------------------------------------------------
__global__ __launch_bounds__(256)
void ssm_scan(const float* __restrict__ xc, const float* __restrict__ dt_in,
              const float* __restrict__ D_in, const float* __restrict__ A_log,
              const float* __restrict__ dt_bias, float* __restrict__ y) {
    const int g  = blockIdx.x;
    const int t  = threadIdx.x;       // 256
    const int hd = t & 127;
    const int sh = t >> 7;            // 0/1 -> states [sh*32, sh*32+32)

    __shared__ float Bsh[DS], Csh[DS], yred[256];

    const float Ag   = -__expf(A_log[g]);
    const float bias = dt_bias[g];
    float hst[32];
#pragma unroll
    for (int j = 0; j < 32; ++j) hst[j] = 0.f;

    for (int l = 0; l < SEQ_L; ++l) {
        const float* row = xc + (size_t)l * D_INNER;
#ifdef HAVE_ASYNC_LDS32
        if (t < DS)          async_ld_b32(row + D_SSM + g * DS + t, &Bsh[t]);
        else if (t < 2 * DS) async_ld_b32(row + D_SSM + NG * DS + g * DS + (t - DS),
                                          &Csh[t - DS]);
        __builtin_amdgcn_s_wait_asynccnt(0);
#else
        if (t < DS)          Bsh[t]      = row[D_SSM + g * DS + t];
        else if (t < 2 * DS) Csh[t - DS] = row[D_SSM + NG * DS + g * DS + (t - DS)];
#endif
        __syncthreads();

        float dtv = dt_in[(size_t)l * NG + g] + bias;
        dtv = (dtv > 20.f) ? dtv : log1pf(__expf(dtv));     // softplus
        const float dA  = __expf(dtv * Ag);
        const float xv  = row[g * HD + hd];
        const float dtx = dtv * xv;
        float yp = 0.f;
#pragma unroll
        for (int j = 0; j < 32; ++j) {
            int s = sh * 32 + j;
            hst[j] = dA * hst[j] + dtx * Bsh[s];
            yp += hst[j] * Csh[s];
        }
        yred[t] = yp;
        __syncthreads();
        if (sh == 0)
            y[(size_t)l * D_SSM + g * HD + hd] =
                yred[t] + yred[t + 128] + D_in[(size_t)l * NG + g] * xv;
    }
}

// ---------------------------------------------------------------------------
// per-head RMSNorm (HD=128) + SiLU(z) gate -> gated [L, D_SSM] in bf16
// ---------------------------------------------------------------------------
__global__ __launch_bounds__(128)
void rms_gate(const float* __restrict__ y, const float* __restrict__ z,
              const float* __restrict__ nw, uint16_t* __restrict__ gated) {
    const int g = blockIdx.x, l = blockIdx.y, t = threadIdx.x;   // t = hd
    const int lane = t & 31, wid = t >> 5;
    __shared__ float red[4];
    const size_t idx = (size_t)l * D_SSM + g * HD + t;
    float v  = y[idx];
    float ss = v * v;
#pragma unroll
    for (int off = 16; off; off >>= 1) ss += __shfl_xor(ss, off, 32);
    if (lane == 0) red[wid] = ss;
    __syncthreads();
    float tot = red[0] + red[1] + red[2] + red[3];
    float yn  = v * rsqrtf(tot * (1.f / HD) + EPS_RMS) * nw[t];
    float zv  = z[idx];
    gated[idx] = bf16_of(yn * zv * sigmoidf_(zv));
}

// ---------------------------------------------------------------------------
// launch
// ---------------------------------------------------------------------------
extern "C" void kernel_launch(void* const* d_in, const int* in_sizes, int n_in,
                              void* d_out, int out_size, void* d_ws, size_t ws_size,
                              hipStream_t stream) {
    const float* h      = (const float*)d_in[0];   // [L, HID]
    const float* W_qkv  = (const float*)d_in[1];   // [D_INNER, HID]
    const float* W_z    = (const float*)d_in[2];   // [D_SSM, HID]
    const float* W_a    = (const float*)d_in[3];   // [NG, HID]
    const float* W_b    = (const float*)d_in[4];   // [NG, HID]
    const float* conv_w = (const float*)d_in[5];   // [D_INNER, 1, K]
    const float* W_out  = (const float*)d_in[6];   // [HID, D_SSM]
    const float* norm_w = (const float*)d_in[7];   // [HD]
    const float* A_log  = (const float*)d_in[8];   // [NG]
    const float* dt_b   = (const float*)d_in[9];   // [NG]
    float* out = (float*)d_out;                    // [L, HID]

    char* wsb = (char*)d_ws;
    // f32 region
    float* xBC   = (float*)(wsb);                   // 33554432 B [L,D_INNER] (dead after conv)
    float* xconv = (float*)(wsb + 33554432);        // 33554432 B [L,D_INNER]
    float* zbuf  = (float*)(wsb + 67108864);        // 16777216 B [L,D_SSM]
    float* ybuf  = (float*)(wsb + 83886080);        // 16777216 B [L,D_SSM]
    float* dtbuf = (float*)(wsb + 100663296);       //   131072 B [L,NG]
    float* Dbuf  = (float*)(wsb + 100794368);       //   131072 B [L,NG]
    // bf16 region (persistent)
    uint16_t* hb    = (uint16_t*)(wsb + 100925440); //  5242880 B [L,HID]
    uint16_t* Wqkvb = (uint16_t*)(wsb + 106168320); // 41943040 B [D_INNER,HID]
    uint16_t* Wzb   = (uint16_t*)(wsb + 148111360); // 20971520 B [D_SSM,HID]
    // bf16 aliases inside dead xBC region (valid only after conv_silu)
    uint16_t* Woutb  = (uint16_t*)(wsb);            // 20971520 B [HID,D_SSM]
    uint16_t* gatedb = (uint16_t*)(wsb + 20971520); //  8388608 B [L,D_SSM]

    // 0) f32 -> bf16 pre-pass (bandwidth-bound; weights stay hot in 192MB L2)
    {
        int n4;
        n4 = SEQ_L * HID / 4;
        cvt_f32_bf16<<<(n4 + 255) / 256, 256, 0, stream>>>(h, (uint32_t*)hb, n4);
        n4 = D_INNER * HID / 4;
        cvt_f32_bf16<<<(n4 + 255) / 256, 256, 0, stream>>>(W_qkv, (uint32_t*)Wqkvb, n4);
        n4 = D_SSM * HID / 4;
        cvt_f32_bf16<<<(n4 + 255) / 256, 256, 0, stream>>>(W_z, (uint32_t*)Wzb, n4);
    }
    // 1) xBC = h @ W_qkv.T
    gemm_bf16_wmma<<<dim3(D_INNER / 128, SEQ_L / 256), 256, 0, stream>>>(
        hb, Wqkvb, xBC, SEQ_L, D_INNER, HID);
    // 2) z = h @ W_z.T
    gemm_bf16_wmma<<<dim3(D_SSM / 128, SEQ_L / 256), 256, 0, stream>>>(
        hb, Wzb, zbuf, SEQ_L, D_SSM, HID);
    // 3) dt = h @ W_b.T, D = h @ W_a.T
    proj_small<<<SEQ_L, 256, 0, stream>>>(h, W_a, W_b, dtbuf, Dbuf);
    // 4) causal depthwise conv + SiLU  (last reader of xBC)
    conv_silu<<<dim3(D_INNER / 256, SEQ_L), 256, 0, stream>>>(xBC, conv_w, xconv);
    // 4b) W_out -> bf16 into the now-dead xBC region
    {
        int n4 = HID * D_SSM / 4;
        cvt_f32_bf16<<<(n4 + 255) / 256, 256, 0, stream>>>(W_out, (uint32_t*)Woutb, n4);
    }
    // 5) SSM scan
    ssm_scan<<<NG, 256, 0, stream>>>(xconv, dtbuf, Dbuf, A_log, dt_b, ybuf);
    // 6) RMSNorm + gate -> bf16
    rms_gate<<<dim3(NG, SEQ_L), 128, 0, stream>>>(ybuf, zbuf, norm_w, gatedb);
    // 7) out = gated @ W_out.T
    gemm_bf16_wmma<<<dim3(HID / 128, SEQ_L / 256), 256, 0, stream>>>(
        gatedb, Woutb, out, SEQ_L, HID, D_SSM);
}
